// NgpNet_6184752906843
// MI455X (gfx1250) — compile-verified, matrix operand order checked by reading, not run on hardware
//
#include <hip/hip_runtime.h>
#include <hip/hip_bf16.h>

#define S_SAMPLES 200
#define THASH     (1u << 19)
#define HMASK     (THASH - 1u)

typedef __attribute__((ext_vector_type(16))) _Float16 v16h;
typedef __attribute__((ext_vector_type(8)))  _Float16 v8h;
typedef __attribute__((ext_vector_type(4)))  _Float16 v4h;
typedef __attribute__((ext_vector_type(8)))  float    v8f;

// Packed-weight geometry: each chunk = one WMMA B fragment for a whole wave:
// 32 lanes x 16 halves = 512 f16 = 1KB. Lane data contiguous (16 halves).
//   L1: chunks  0..15  (kc 0..3  x nt 0..3)
//   L2: chunks 16..23  (kc 0..1  x nt 0..3)
//   L3: chunks 24..31  (kc 0..1  x nt 0..3)
//   L4: chunks 32..33  (kc 0..1,  nt 0)
#define NCHUNKS   34
#define CHUNK_ELT 512

// ---------------------------------------------------------------------------
// Hash-grid trilinear interpolation, D=1 (label tables)
// ---------------------------------------------------------------------------
__device__ __forceinline__ float hash_interp1(const float* __restrict__ tbl,
                                              float px, float py, float pz, int res) {
  float xf = px * (float)res, yf = py * (float)res, zf = pz * (float)res;
  float fx = floorf(xf), fy = floorf(yf), fz = floorf(zf);
  int x0 = (int)fx, y0 = (int)fy, z0 = (int)fz;
  float wx = xf - fx, wy = yf - fy, wz = zf - fz;
  float acc = 0.f;
#pragma unroll
  for (int c = 0; c < 8; ++c) {
    int cx = (c >> 2) & 1, cy = (c >> 1) & 1, cz = c & 1;
    unsigned hx = (unsigned)(x0 + cx);                    // prime 1
    unsigned hy = (unsigned)(y0 + cy) * 2654435761u;
    unsigned hz = (unsigned)(z0 + cz) * 805459861u;
    unsigned idx = (hx ^ hy ^ hz) & HMASK;
    float w = (cx ? wx : 1.f - wx) * (cy ? wy : 1.f - wy) * (cz ? wz : 1.f - wz);
    acc = fmaf(w, tbl[idx], acc);
  }
  return acc;
}

// ---------------------------------------------------------------------------
// Kernel 0: pre-swizzle fp32 weights into WMMA B-fragment f16 layout.
// One thread per packed element; zero-pads K (120->128) and N (3->16).
// ---------------------------------------------------------------------------
__global__ void __launch_bounds__(256)
pack_weights(const float* __restrict__ W1, const float* __restrict__ W2,
             const float* __restrict__ W3, const float* __restrict__ W4,
             _Float16* __restrict__ wp) {
  int gid = blockIdx.x * blockDim.x + threadIdx.x;
  if (gid >= NCHUNKS * CHUNK_ELT) return;
  int chunk  = gid >> 9;
  int within = gid & 511;
  int lane   = within >> 4;   // 0..31
  int e      = within & 15;   // 0..15

  const float* W;
  int kvalid, nvalid, nstride, kc, nt;
  if (chunk < 16)      { int c = chunk;      W = W1; kvalid = 120; nvalid = 64; nstride = 64; kc = c >> 2; nt = c & 3; }
  else if (chunk < 24) { int c = chunk - 16; W = W2; kvalid = 64;  nvalid = 64; nstride = 64; kc = c >> 2; nt = c & 3; }
  else if (chunk < 32) { int c = chunk - 24; W = W3; kvalid = 64;  nvalid = 64; nstride = 64; kc = c >> 2; nt = c & 3; }
  else                 { int c = chunk - 32; W = W4; kvalid = 64;  nvalid = 3;  nstride = 3;  kc = c;      nt = 0;     }

  int n  = nt * 16 + (lane & 15);
  int kk = kc * 32 + ((lane & 16) ? 8 : 0);
  int k  = kk + e + ((e >= 8) ? 8 : 0);   // e<8 -> kk+e ; e>=8 -> kk+16+(e-8)

  float v = (k < kvalid && n < nvalid) ? W[k * nstride + n] : 0.f;
  wp[gid] = (_Float16)v;
}

// ---------------------------------------------------------------------------
// Kernel 1: label field, per-ray max (hits) and first-hit index.
// One block (256 threads) per ray; 200 active sample threads.
// ---------------------------------------------------------------------------
__global__ void __launch_bounds__(256)
label_kernel(const float* __restrict__ x, const float* __restrict__ table_label,
             const float* __restrict__ w_label, const float* __restrict__ b_label,
             float* __restrict__ out_hits, float* __restrict__ out_label,
             int* __restrict__ first_idx, int B) {
  int b   = blockIdx.x;
  int tid = threadIdx.x;
  __shared__ float rmax[256];
  __shared__ int   rmin[256];

  float lbl  = -1.f;
  int   cand = 0x7fffffff;
  float sx = x[b * 4 + 0], sy = x[b * 4 + 1];
  float ex = x[b * 4 + 2], ey = x[b * 4 + 3];

  if (tid < S_SAMPLES) {
    float t  = (float)tid * (1.f / (float)(S_SAMPLES - 1));
    float px = sx * (1.f - t) + ex * t;
    float py = sy * (1.f - t) + ey * t;
    float pz = t;
    float logit = b_label[0];
#pragma unroll
    for (int l = 0; l < 5; ++l) {
      int res = 8 << l;
      logit = fmaf(hash_interp1(table_label + (size_t)l * THASH, px, py, pz, res),
                   w_label[l], logit);
    }
    lbl = 1.f / (1.f + __expf(-logit));
    out_label[(size_t)b * S_SAMPLES + tid] = lbl;
    if (lbl > 0.5f) cand = tid;
  }
  rmax[tid] = lbl;
  rmin[tid] = cand;
  __syncthreads();
#pragma unroll
  for (int s = 128; s > 0; s >>= 1) {
    if (tid < s) {
      rmax[tid] = fmaxf(rmax[tid], rmax[tid + s]);
      rmin[tid] = min(rmin[tid], rmin[tid + s]);
    }
    __syncthreads();
  }
  if (tid == 0) {
    out_hits[b]  = rmax[0];
    first_idx[b] = (rmin[0] == 0x7fffffff) ? 0 : rmin[0];
  }
}

// ---------------------------------------------------------------------------
// Kernel 2: rgb hash-grid features ONLY at the 5 gathered samples per ray
// (40x fewer gathers than the reference). One thread per (ray, offset, level).
// Writes f16 feature matrix [B, 128] (cols 120..127 zero K-padding).
// ---------------------------------------------------------------------------
__global__ void __launch_bounds__(256)
feat_kernel(const float* __restrict__ x, const float* __restrict__ table_rgb,
            const int* __restrict__ first_idx, _Float16* __restrict__ feat, int B) {
  int gid = blockIdx.x * blockDim.x + threadIdx.x;
  if (gid >= B * 30) return;
  int b = gid / 30;
  int r = gid % 30;
  int o = r / 6;   // offset slot 0..4  -> first + (o-2)
  int l = r % 6;   // level 0..5

  int s = first_idx[b] + o - 2;
  s = min(max(s, 0), S_SAMPLES - 1);
  float t  = (float)s * (1.f / (float)(S_SAMPLES - 1));
  float px = x[b * 4 + 0] * (1.f - t) + x[b * 4 + 2] * t;
  float py = x[b * 4 + 1] * (1.f - t) + x[b * 4 + 3] * t;
  float pz = t;

  int res = 4 << l;
  float xf = px * (float)res, yf = py * (float)res, zf = pz * (float)res;
  float fx = floorf(xf), fy = floorf(yf), fz = floorf(zf);
  int x0 = (int)fx, y0 = (int)fy, z0 = (int)fz;
  float wx = xf - fx, wy = yf - fy, wz = zf - fz;

  const float4* __restrict__ tbl = (const float4*)table_rgb + (size_t)l * THASH;
  float a0 = 0.f, a1 = 0.f, a2 = 0.f, a3 = 0.f;
#pragma unroll
  for (int c = 0; c < 8; ++c) {
    int cx = (c >> 2) & 1, cy = (c >> 1) & 1, cz = c & 1;
    unsigned hx = (unsigned)(x0 + cx);
    unsigned hy = (unsigned)(y0 + cy) * 2654435761u;
    unsigned hz = (unsigned)(z0 + cz) * 805459861u;
    unsigned idx = (hx ^ hy ^ hz) & HMASK;
    float w = (cx ? wx : 1.f - wx) * (cy ? wy : 1.f - wy) * (cz ? wz : 1.f - wz);
    float4 f = tbl[idx];
    a0 = fmaf(w, f.x, a0); a1 = fmaf(w, f.y, a1);
    a2 = fmaf(w, f.z, a2); a3 = fmaf(w, f.w, a3);
  }
  v4h out4 = { (_Float16)a0, (_Float16)a1, (_Float16)a2, (_Float16)a3 };
  *(v4h*)(feat + (size_t)b * 128 + o * 24 + l * 4) = out4;   // 8B-aligned
  if (r == 0) {  // zero K-padding columns 120..127 once per ray
    v8h z = {};
    *(v8h*)(feat + (size_t)b * 128 + 120) = z;               // 16B-aligned
  }
}

// ---------------------------------------------------------------------------
// WMMA fragment loads. A (16xK row-major f16 source, ISA 7.12.2 layout):
// lanes 0-15 -> row=lane, K = k0..k0+7 | k0+16..k0+23; lanes 16-31 -> +8 halves.
// B comes pre-swizzled: 16 contiguous halves per lane.
// ---------------------------------------------------------------------------
__device__ __forceinline__ v16h load_a_rowmajor(const _Float16* base, int ld,
                                                int lane, int k0) {
  int row = lane & 15;
  int kk  = k0 + ((lane & 16) ? 8 : 0);
  const _Float16* p = base + row * ld + kk;
  v16h a;
  *reinterpret_cast<v8h*>(&a)       = *reinterpret_cast<const v8h*>(p);
  *(reinterpret_cast<v8h*>(&a) + 1) = *reinterpret_cast<const v8h*>(p + 16);
  return a;
}

__device__ __forceinline__ v16h load_b_packed(const _Float16* __restrict__ wp,
                                              int lane) {
  const v8h* p = reinterpret_cast<const v8h*>(wp + lane * 16);
  v16h b;
  *reinterpret_cast<v8h*>(&b)       = p[0];
  *(reinterpret_cast<v8h*>(&b) + 1) = p[1];
  return b;
}

// ---------------------------------------------------------------------------
// Kernel 3: fused MLP 120(->128) -> 64 -> 64 -> 64 -> 3 via v_wmma_f32_16x16x32_f16.
// 1 block = 128 threads = 4 waves; block handles 16 rays; wave w owns N-tile w.
// Activations ping-pong through LDS as f16. EXEC is all-ones at every WMMA.
// ---------------------------------------------------------------------------
__global__ void __launch_bounds__(128)
mlp_kernel(const _Float16* __restrict__ feat, const _Float16* __restrict__ wp,
           const float* __restrict__ b1, const float* __restrict__ b2,
           const float* __restrict__ b3, const float* __restrict__ b4,
           float* __restrict__ out_rgb, int B) {
  __shared__ _Float16 h1[16 * 64];
  __shared__ _Float16 h2[16 * 64];

  int lane  = threadIdx.x & 31;
  int wave  = threadIdx.x >> 5;
  int r0    = blockIdx.x * 16;
  int n0    = wave * 16;
  int nloc  = lane & 15;
  int mbase = (lane & 16) ? 8 : 0;

  // ---- Layer 1: feat[16x128] x W1 ----
  {
    v8f c = {};
#pragma unroll
    for (int kc = 0; kc < 4; ++kc) {
      v16h a  = load_a_rowmajor(feat + (size_t)r0 * 128, 128, lane, kc * 32);
      v16h bm = load_b_packed(wp + (size_t)(kc * 4 + wave) * CHUNK_ELT, lane);
      c = __builtin_amdgcn_wmma_f32_16x16x32_f16(false, a, false, bm, (short)0, c,
                                                 false, false);
    }
    float bias = b1[n0 + nloc];
#pragma unroll
    for (int i = 0; i < 8; ++i) {
      float v = c[i] + bias;
      h1[(mbase + i) * 64 + n0 + nloc] = (_Float16)(v > 0.f ? v : 0.f);
    }
  }
  __syncthreads();

  // ---- Layer 2: h1 x W2 -> h2 ----
  {
    v8f c = {};
#pragma unroll
    for (int kc = 0; kc < 2; ++kc) {
      v16h a  = load_a_rowmajor(h1, 64, lane, kc * 32);
      v16h bm = load_b_packed(wp + (size_t)(16 + kc * 4 + wave) * CHUNK_ELT, lane);
      c = __builtin_amdgcn_wmma_f32_16x16x32_f16(false, a, false, bm, (short)0, c,
                                                 false, false);
    }
    float bias = b2[n0 + nloc];
#pragma unroll
    for (int i = 0; i < 8; ++i) {
      float v = c[i] + bias;
      h2[(mbase + i) * 64 + n0 + nloc] = (_Float16)(v > 0.f ? v : 0.f);
    }
  }
  __syncthreads();

  // ---- Layer 3: h2 x W3 -> h1 ----
  {
    v8f c = {};
#pragma unroll
    for (int kc = 0; kc < 2; ++kc) {
      v16h a  = load_a_rowmajor(h2, 64, lane, kc * 32);
      v16h bm = load_b_packed(wp + (size_t)(24 + kc * 4 + wave) * CHUNK_ELT, lane);
      c = __builtin_amdgcn_wmma_f32_16x16x32_f16(false, a, false, bm, (short)0, c,
                                                 false, false);
    }
    float bias = b3[n0 + nloc];
#pragma unroll
    for (int i = 0; i < 8; ++i) {
      float v = c[i] + bias;
      h1[(mbase + i) * 64 + n0 + nloc] = (_Float16)(v > 0.f ? v : 0.f);
    }
  }
  __syncthreads();

  // ---- Layer 4: h1 x W4 -> rgb (wave 0 only; wave-uniform branch) ----
  if (wave == 0) {
    v8f c = {};
#pragma unroll
    for (int kc = 0; kc < 2; ++kc) {
      v16h a  = load_a_rowmajor(h1, 64, lane, kc * 32);
      v16h bm = load_b_packed(wp + (size_t)(32 + kc) * CHUNK_ELT, lane);
      c = __builtin_amdgcn_wmma_f32_16x16x32_f16(false, a, false, bm, (short)0, c,
                                                 false, false);
    }
    if (nloc < 3) {
      float bias = b4[nloc];
#pragma unroll
      for (int i = 0; i < 8; ++i) {
        int row = r0 + mbase + i;
        if (row < B) out_rgb[(size_t)row * 3 + nloc] = c[i] + bias;
      }
    }
  }
}

// ---------------------------------------------------------------------------
// Launch: out = [hits(B) | label(B*200) | rgb(B*3)]
// ws     = [first(B ints) | feat f16 (B*128) | packed weights f16 (34*512)]
// ---------------------------------------------------------------------------
extern "C" void kernel_launch(void* const* d_in, const int* in_sizes, int n_in,
                              void* d_out, int out_size, void* d_ws, size_t ws_size,
                              hipStream_t stream) {
  const float* x           = (const float*)d_in[0];
  const float* table_label = (const float*)d_in[1];
  const float* table_rgb   = (const float*)d_in[2];
  const float* w_label     = (const float*)d_in[3];
  const float* b_label     = (const float*)d_in[4];
  const float* W1 = (const float*)d_in[5];
  const float* b1 = (const float*)d_in[6];
  const float* W2 = (const float*)d_in[7];
  const float* b2 = (const float*)d_in[8];
  const float* W3 = (const float*)d_in[9];
  const float* b3 = (const float*)d_in[10];
  const float* W4 = (const float*)d_in[11];
  const float* b4 = (const float*)d_in[12];

  int B = in_sizes[0] / 4;

  float* out_hits  = (float*)d_out;
  float* out_label = out_hits + B;
  float* out_rgb   = out_label + (size_t)B * S_SAMPLES;

  int*      first = (int*)d_ws;
  _Float16* feat  = (_Float16*)((char*)d_ws + (size_t)B * sizeof(int));
  _Float16* wp    = feat + (size_t)B * 128;

  pack_weights<<<(NCHUNKS * CHUNK_ELT + 255) / 256, 256, 0, stream>>>(W1, W2, W3,
                                                                      W4, wp);

  label_kernel<<<B, 256, 0, stream>>>(x, table_label, w_label, b_label,
                                      out_hits, out_label, first, B);

  int feat_threads = B * 30;
  feat_kernel<<<(feat_threads + 255) / 256, 256, 0, stream>>>(x, table_rgb, first,
                                                              feat, B);

  mlp_kernel<<<(B + 15) / 16, 128, 0, stream>>>(feat, wp, b1, b2, b3, b4,
                                                out_rgb, B);
}
